// GraphEdgeActionGNN_4020089389507
// MI455X (gfx1250) — matrix operand order repaired and checked by reading, not compiled, not run on hardware
//
#include <hip/hip_runtime.h>
#include <cstddef>

// ---------------------------------------------------------------------------
// Problem constants (match reference)
// ---------------------------------------------------------------------------
constexpr int    BATCH    = 256;
constexpr int    NNODES   = 512;
constexpr int    D        = 128;
constexpr long   NTOTAL   = (long)BATCH * NNODES;      // 131072
constexpr long   NEDGES   = NTOTAL * 8;                // 1048576
constexpr int    EDGESDIM = NNODES * (NNODES - 1) / 2; // 130816
// d_out layout: action_type[256*3] | edge_class[256*8] | node_class[256*1] | edge_actions[256*130816]
constexpr size_t OUT_EDGE_OFF = 256 * 3 + 256 * 8 + 256 * 1; // 3072

typedef _Float16 v16h_t __attribute__((ext_vector_type(16)));
typedef _Float16 v8h_t  __attribute__((ext_vector_type(8)));
typedef _Float16 v2h_t  __attribute__((ext_vector_type(2)));
typedef float    v8f_t  __attribute__((ext_vector_type(8)));

// ---------------------------------------------------------------------------
// 1. Gather embeddings (x = emb[nf]) and zero the aggregation buffer
// ---------------------------------------------------------------------------
__global__ __launch_bounds__(256) void gather_zero_kernel(
    const int* __restrict__ nf, const float* __restrict__ emb,
    float* __restrict__ x, float* __restrict__ agg)
{
    size_t gid = (size_t)blockIdx.x * blockDim.x + threadIdx.x; // NTOTAL*32 threads
    size_t i = gid >> 5;
    int    c = (int)(gid & 31);
    const float4* e4 = (const float4*)emb;
    ((float4*)x)[i * 32 + c]   = e4[(size_t)nf[i] * 32 + c];
    ((float4*)agg)[i * 32 + c] = make_float4(0.f, 0.f, 0.f, 0.f);
}

// ---------------------------------------------------------------------------
// 2. Edge scatter: agg[dst] += x[src]  (one wave per edge, float4 per lane)
// ---------------------------------------------------------------------------
__global__ __launch_bounds__(256) void edge_scatter_kernel(
    const int* __restrict__ ei, const float* __restrict__ x, float* __restrict__ agg)
{
    size_t tid = (size_t)blockIdx.x * blockDim.x + threadIdx.x; // NEDGES*32 threads
    size_t e   = tid >> 5;
    int    lane = (int)(tid & 31);
    int s = ei[e];
    int d = ei[(size_t)NEDGES + e];
    float4 v = ((const float4*)x)[(size_t)s * 32 + lane];
    float* ap = agg + (size_t)d * D + lane * 4;
    atomicAdd(ap + 0, v.x);
    atomicAdd(ap + 1, v.y);
    atomicAdd(ap + 2, v.z);
    atomicAdd(ap + 3, v.w);
}

// ---------------------------------------------------------------------------
// 3. (x + agg) -> f16
// ---------------------------------------------------------------------------
__global__ __launch_bounds__(256) void add_convert_kernel(
    const float* __restrict__ x, const float* __restrict__ agg, _Float16* __restrict__ out)
{
    size_t gid = (size_t)blockIdx.x * blockDim.x + threadIdx.x; // NTOTAL*64 threads
    float2 a = ((const float2*)x)[gid];
    float2 b = ((const float2*)agg)[gid];
    v2h_t h;
    h[0] = (_Float16)(a.x + b.x);
    h[1] = (_Float16)(a.y + b.y);
    ((v2h_t*)out)[gid] = h;
}

// ---------------------------------------------------------------------------
// 4. Weight convert + transpose: Wt[n][k] = (f16) W[k][n]   (128x128)
//    Transposed storage makes the WMMA B-fragment a single contiguous
//    32-byte load per lane (K runs along memory).
// ---------------------------------------------------------------------------
__global__ __launch_bounds__(256) void wconv_t_kernel(
    const float* __restrict__ in, _Float16* __restrict__ out)
{
    int i = blockIdx.x * blockDim.x + threadIdx.x; // 16384 threads
    int n = i >> 7;
    int k = i & 127;
    out[i] = (_Float16)in[k * 128 + n];
}

// ---------------------------------------------------------------------------
// 5. Fused WMMA GEMM: out = epilogue(A[Mx128] @ W[128x128] + bias)
//    Wt is the transposed f16 weight (N-major). No LDS; per-wave 16-row strip.
//    A frag (16-bit 16x32): lane l -> row l&15, K = 16*(l>=16) + {0..15}
//                           => two contiguous v8h loads.
//    B frag (16-bit 32x16): lane l -> col l&15, K = 16*(l>=16) + {0..15}
//                           => one contiguous v16h load from Wt.
//    C tile (f32): VGPR v -> row v + 8*(l>=16), col l&15.
//    LayerNorm epilogue: row stats fully in registers via __shfl_xor over the
//    16-lane half-group (wave32), no LDS, no barriers. All store offsets are
//    32-bit so stores use the compact saddr+voffset scale_offset form.
// ---------------------------------------------------------------------------
template <bool DO_LN, bool DO_RELU, bool OUT32>
__global__ __launch_bounds__(256) void gemm_ln_kernel(
    const _Float16* __restrict__ A, const _Float16* __restrict__ Wt,
    const float* __restrict__ bias, const float* __restrict__ g,
    const float* __restrict__ beta, _Float16* __restrict__ out16,
    float* __restrict__ out32)
{
    const int lane = threadIdx.x & 31;
    const int wv   = threadIdx.x >> 5;
    const int col  = lane & 15;
    const int hi   = lane >> 4;
    const int m0   = (blockIdx.x * 8 + wv) * 16;   // fits 32-bit (<= 131072)

    v8f_t acc[8];
#pragma unroll
    for (int nt = 0; nt < 8; ++nt) acc[nt] = (v8f_t)(0.0f);

    const _Float16* Arow = A + (size_t)(m0 + col) * 128;
#pragma unroll
    for (int s = 0; s < 4; ++s) {
        const _Float16* Ap = Arow + s * 32 + hi * 16;
        v8h_t alo = *(const v8h_t*)(Ap);        // K window +0..7
        v8h_t ahi = *(const v8h_t*)(Ap + 8);    // K window +8..15
        v16h_t a = __builtin_shufflevector(alo, ahi,
            0, 1, 2, 3, 4, 5, 6, 7, 8, 9, 10, 11, 12, 13, 14, 15);
#pragma unroll
        for (int nt = 0; nt < 8; ++nt) {
            const _Float16* Wp = Wt + (nt * 16 + col) * 128 + s * 32 + hi * 16;
            v16h_t b = *(const v16h_t*)Wp;      // contiguous, 32B aligned
            acc[nt] = __builtin_amdgcn_wmma_f32_16x16x32_f16(
                false, a, false, b, (short)0, acc[nt], false, false);
        }
    }

    float bcol[8];
#pragma unroll
    for (int nt = 0; nt < 8; ++nt) bcol[nt] = bias[nt * 16 + col];

    if (!DO_LN) {
#pragma unroll
        for (int nt = 0; nt < 8; ++nt) {
#pragma unroll
            for (int v = 0; v < 8; ++v) {
                float val = acc[nt][v] + bcol[nt];
                if (DO_RELU) val = fmaxf(val, 0.0f);
                int off = (m0 + v + hi * 8) * 128 + nt * 16 + col; // 32-bit offset
                out16[off] = (_Float16)val;
                if (OUT32) out32[off] = val;
            }
        }
    } else {
        float gcol[8], becol[8];
#pragma unroll
        for (int nt = 0; nt < 8; ++nt) {
            gcol[nt]  = g[nt * 16 + col];
            becol[nt] = beta[nt * 16 + col];
        }
#pragma unroll
        for (int v = 0; v < 8; ++v) {
            float vals[8];
            float s0 = 0.f, s1 = 0.f;
#pragma unroll
            for (int nt = 0; nt < 8; ++nt) {
                vals[nt] = acc[nt][v] + bcol[nt];
                s0 += vals[nt];
                s1 += vals[nt] * vals[nt];
            }
            // reduce across the 16 lanes holding this row (xor masks stay in group)
#pragma unroll
            for (int m = 1; m < 16; m <<= 1) {
                s0 += __shfl_xor(s0, m, 32);
                s1 += __shfl_xor(s1, m, 32);
            }
            float mean = s0 * (1.0f / 128.0f);
            float var  = s1 * (1.0f / 128.0f) - mean * mean;
            float rstd = rsqrtf(var + 1e-5f);
            int rowoff = (m0 + v + hi * 8) * 128 + col;            // 32-bit offset
#pragma unroll
            for (int nt = 0; nt < 8; ++nt) {
                float val = (vals[nt] - mean) * rstd * gcol[nt] + becol[nt];
                if (DO_RELU) val = fmaxf(val, 0.0f);
                int off = rowoff + nt * 16;
                out16[off] = (_Float16)val;
                if (OUT32) out32[off] = val;
            }
        }
    }
}

// ---------------------------------------------------------------------------
// 6. Per-batch mean of x (512 nodes) -> f16 for the exit GEMM
// ---------------------------------------------------------------------------
__global__ __launch_bounds__(128) void means_kernel(
    const float* __restrict__ xf32, _Float16* __restrict__ means16)
{
    int b = blockIdx.x;
    int d = threadIdx.x;
    const float* p = xf32 + (size_t)b * NNODES * D;
    float s = 0.f;
    for (int n = 0; n < NNODES; ++n) s += p[(size_t)n * D + d];
    means16[(size_t)b * D + d] = (_Float16)(s * (1.0f / (float)NNODES));
}

// ---------------------------------------------------------------------------
// 7. Final exit dot-product + assemble small outputs
// ---------------------------------------------------------------------------
__global__ __launch_bounds__(256) void exit_final_kernel(
    const _Float16* __restrict__ exith, const float* __restrict__ W2,
    const float* __restrict__ b2, float* __restrict__ out)
{
    int b = blockIdx.x * blockDim.x + threadIdx.x;
    if (b >= BATCH) return;
    float s = 0.f;
    for (int d = 0; d < D; ++d) s += (float)exith[(size_t)b * D + d] * W2[d];
    s += b2[0];
    out[b * 3 + 0] = 0.0f;          // action_type col 0
    out[b * 3 + 1] = 1.0f - s;      // action_type col 1
    out[b * 3 + 2] = s;             // action_type col 2
    for (int j = 0; j < 8; ++j) out[256 * 3 + b * 8 + j] = 0.0f; // edge_class
    out[256 * 3 + 256 * 8 + b] = 0.0f;                           // node_class
}

// ---------------------------------------------------------------------------
// 8. Batched scores: S = Xb @ Xb^T / sqrt(D), packed upper-triangle output.
//    One wave per (batch, row-tile); A fragments loaded once, reused across
//    all j-tiles. Diagonal tile peeled so the 31-iteration main loop has
//    branch-free stores; B fragments preloaded as one clause. All packed-triu
//    offsets are 32-bit (max 33.5M) for compact scale_offset addressing.
// ---------------------------------------------------------------------------
__global__ __launch_bounds__(32) void scores_kernel(
    const _Float16* __restrict__ xf16, float* __restrict__ out_edges)
{
    const int ti   = blockIdx.x;
    const int b    = blockIdx.y;
    const int lane = threadIdx.x;
    const int col  = lane & 15;
    const int hi   = lane >> 4;
    const size_t base = (size_t)b * NNODES;
    constexpr float SCALE = 0.08838834764831845f; // 1/sqrt(128)

    // A fragments for this 16-row strip (reused for all column tiles)
    v16h_t afr[4];
    const _Float16* Ai = xf16 + (base + (size_t)ti * 16 + col) * 128;
#pragma unroll
    for (int s = 0; s < 4; ++s) {
        const _Float16* Ap = Ai + s * 32 + hi * 16;
        v8h_t alo = *(const v8h_t*)(Ap);
        v8h_t ahi = *(const v8h_t*)(Ap + 8);
        afr[s] = __builtin_shufflevector(alo, ahi,
            0, 1, 2, 3, 4, 5, 6, 7, 8, 9, 10, 11, 12, 13, 14, 15);
    }

    // Packed-triu row bases (32-bit): idx(i,j) = i*(N-1) - i(i+1)/2 + j - 1
    int rowbase[8];
#pragma unroll
    for (int v = 0; v < 8; ++v) {
        int i = ti * 16 + v + hi * 8;
        rowbase[v] = b * EDGESDIM + i * (NNODES - 1) - ((i * (i + 1)) >> 1) - 1;
    }

    // ---- diagonal tile (tj == ti): only j > i elements stored ----
    {
        const _Float16* Bj = xf16 + (base + (size_t)ti * 16 + col) * 128 + hi * 16;
        v16h_t bf0 = *(const v16h_t*)(Bj);
        v16h_t bf1 = *(const v16h_t*)(Bj + 32);
        v16h_t bf2 = *(const v16h_t*)(Bj + 64);
        v16h_t bf3 = *(const v16h_t*)(Bj + 96);
        v8f_t acc = (v8f_t)(0.0f);
        acc = __builtin_amdgcn_wmma_f32_16x16x32_f16(false, afr[0], false, bf0, (short)0, acc, false, false);
        acc = __builtin_amdgcn_wmma_f32_16x16x32_f16(false, afr[1], false, bf1, (short)0, acc, false, false);
        acc = __builtin_amdgcn_wmma_f32_16x16x32_f16(false, afr[2], false, bf2, (short)0, acc, false, false);
        acc = __builtin_amdgcn_wmma_f32_16x16x32_f16(false, afr[3], false, bf3, (short)0, acc, false, false);
        int j = ti * 16 + col;
#pragma unroll
        for (int v = 0; v < 8; ++v) {
            int i = ti * 16 + v + hi * 8;
            if (j > i) out_edges[rowbase[v] + j] = acc[v] * SCALE;
        }
    }

    // ---- off-diagonal tiles: every element is in the upper triangle ----
    for (int tj = ti + 1; tj < 32; ++tj) {
        const _Float16* Bj = xf16 + (base + (size_t)tj * 16 + col) * 128 + hi * 16;
        if (tj + 1 < 32)  // emits global_prefetch_b8 for the next B strip
            __builtin_prefetch(xf16 + (base + (size_t)(tj + 1) * 16 + col) * 128, 0, 1);
        v16h_t bf0 = *(const v16h_t*)(Bj);
        v16h_t bf1 = *(const v16h_t*)(Bj + 32);
        v16h_t bf2 = *(const v16h_t*)(Bj + 64);
        v16h_t bf3 = *(const v16h_t*)(Bj + 96);
        v8f_t acc = (v8f_t)(0.0f);
        acc = __builtin_amdgcn_wmma_f32_16x16x32_f16(false, afr[0], false, bf0, (short)0, acc, false, false);
        acc = __builtin_amdgcn_wmma_f32_16x16x32_f16(false, afr[1], false, bf1, (short)0, acc, false, false);
        acc = __builtin_amdgcn_wmma_f32_16x16x32_f16(false, afr[2], false, bf2, (short)0, acc, false, false);
        acc = __builtin_amdgcn_wmma_f32_16x16x32_f16(false, afr[3], false, bf3, (short)0, acc, false, false);
        int jo = tj * 16 + col;
#pragma unroll
        for (int v = 0; v < 8; ++v)
            out_edges[rowbase[v] + jo] = acc[v] * SCALE;  // unconditional, 32-bit offset
    }
}

// ---------------------------------------------------------------------------
// Host launcher
// ---------------------------------------------------------------------------
extern "C" void kernel_launch(void* const* d_in, const int* in_sizes, int n_in,
                              void* d_out, int out_size, void* d_ws, size_t ws_size,
                              hipStream_t stream)
{
    (void)in_sizes; (void)n_in; (void)out_size; (void)ws_size;

    const int*   nf       = (const int*)d_in[0];
    const int*   ei       = (const int*)d_in[1];
    /* d_in[2] batch_ptr: uniform 512, unused */
    const float* emb      = (const float*)d_in[3];
    const float* gin_W1   = (const float*)d_in[4];
    const float* gin_b1   = (const float*)d_in[5];
    const float* gin_g1   = (const float*)d_in[6];
    const float* gin_be1  = (const float*)d_in[7];
    const float* gin_W2   = (const float*)d_in[8];
    const float* gin_b2   = (const float*)d_in[9];
    const float* post_W1  = (const float*)d_in[10];
    const float* post_b1  = (const float*)d_in[11];
    const float* post_W2  = (const float*)d_in[12];
    const float* post_b2  = (const float*)d_in[13];
    const float* norm_g   = (const float*)d_in[14];
    const float* norm_b   = (const float*)d_in[15];
    const float* exit_W1  = (const float*)d_in[16];
    const float* exit_b1  = (const float*)d_in[17];
    const float* exit_g   = (const float*)d_in[18];
    const float* exit_be  = (const float*)d_in[19];
    const float* exit_W2  = (const float*)d_in[20];
    const float* exit_b2  = (const float*)d_in[21];

    float* out = (float*)d_out;
    char*  ws  = (char*)d_ws;
    const size_t MB = 1ull << 20;

    // Workspace plan (~160 MB + 300 KB), f16 buffers overlaid on retired
    // fp32 regions:
    float*    x      = (float*)(ws + 0);           // 64 MB: gathered emb
    float*    agg    = (float*)(ws + 64 * MB);     // 64 MB: segment sum, later xf32
    _Float16* f16a   = (_Float16*)(ws + 128 * MB); // 32 MB: (x+agg) f16, later p1
    _Float16* hbuf1  = (_Float16*)(ws + 0);        // 32 MB: h1 f16, later xf16
    _Float16* hbuf2  = (_Float16*)(ws + 32 * MB);  // 32 MB: h f16
    float*    xf32   = agg;                        // GEMM4 f32 output
    _Float16* xf16   = hbuf1;                      // GEMM4 f16 output
    char*     tail   = ws + 160 * MB;
    _Float16* w_gin1  = (_Float16*)(tail + 0 * 32768);
    _Float16* w_gin2  = (_Float16*)(tail + 1 * 32768);
    _Float16* w_post1 = (_Float16*)(tail + 2 * 32768);
    _Float16* w_post2 = (_Float16*)(tail + 3 * 32768);
    _Float16* w_exit1 = (_Float16*)(tail + 4 * 32768);
    _Float16* means16 = (_Float16*)(tail + 5 * 32768);
    _Float16* exith16 = (_Float16*)(tail + 5 * 32768 + 65536);

    // 1. gather + zero agg
    gather_zero_kernel<<<(unsigned)(NTOTAL * 32 / 256), 256, 0, stream>>>(nf, emb, x, agg);
    // 2. edge scatter-add
    edge_scatter_kernel<<<(unsigned)(NEDGES * 32 / 256), 256, 0, stream>>>(ei, x, agg);
    // 3. (x+agg) -> f16
    add_convert_kernel<<<(unsigned)(NTOTAL * 64 / 256), 256, 0, stream>>>(x, agg, f16a);
    // 4. weights -> f16 transposed (N-major)
    wconv_t_kernel<<<64, 256, 0, stream>>>(gin_W1,  w_gin1);
    wconv_t_kernel<<<64, 256, 0, stream>>>(gin_W2,  w_gin2);
    wconv_t_kernel<<<64, 256, 0, stream>>>(post_W1, w_post1);
    wconv_t_kernel<<<64, 256, 0, stream>>>(post_W2, w_post2);
    wconv_t_kernel<<<64, 256, 0, stream>>>(exit_W1, w_exit1);

    const unsigned nodeBlocks = (unsigned)(NTOTAL / 16 / 8); // 1024
    // 5. h1 = relu(LN(xin @ gin_W1 + b1))
    gemm_ln_kernel<true, true, false><<<nodeBlocks, 256, 0, stream>>>(
        f16a, w_gin1, gin_b1, gin_g1, gin_be1, hbuf1, nullptr);
    // 6. h = h1 @ gin_W2 + b2
    gemm_ln_kernel<false, false, false><<<nodeBlocks, 256, 0, stream>>>(
        hbuf1, w_gin2, gin_b2, nullptr, nullptr, hbuf2, nullptr);
    // 7. p1 = relu(h @ post_W1 + post_b1)
    gemm_ln_kernel<false, true, false><<<nodeBlocks, 256, 0, stream>>>(
        hbuf2, w_post1, post_b1, nullptr, nullptr, f16a, nullptr);
    // 8. x = LN(p1 @ post_W2 + post_b2)  -> both f32 and f16
    gemm_ln_kernel<true, false, true><<<nodeBlocks, 256, 0, stream>>>(
        f16a, w_post2, post_b2, norm_g, norm_b, xf16, xf32);

    // 9. per-batch means
    means_kernel<<<BATCH, 128, 0, stream>>>(xf32, means16);
    // 10. exit hidden = relu(LN(means @ exit_W1 + exit_b1))
    gemm_ln_kernel<true, true, false><<<BATCH / 16 / 8, 256, 0, stream>>>(
        means16, w_exit1, exit_b1, exit_g, exit_be, exith16, nullptr);
    // 11. exit dot + small outputs
    exit_final_kernel<<<1, 256, 0, stream>>>(exith16, exit_W2, exit_b2, out);

    // 12. batched upper-tri scores
    scores_kernel<<<dim3(NNODES / 16, BATCH), 32, 0, stream>>>(xf16, out + OUT_EDGE_OFF);
}
